// MultiAttrEncoder_42597485642028
// MI455X (gfx1250) — matrix operand order repaired and checked by reading, not compiled, not run on hardware
//
#include <hip/hip_runtime.h>
#include <hip/hip_bf16.h>

// ---------------------------------------------------------------------------
// Types for CDNA5 WMMA (wave32): A/B = 16 x bf16 (8 VGPRs), C/D = 8 x f32.
// ---------------------------------------------------------------------------
typedef __bf16 bf16_t;
typedef __attribute__((ext_vector_type(16))) __bf16 v16bf;
typedef __attribute__((ext_vector_type(8)))  __bf16 v8bf;
typedef __attribute__((ext_vector_type(8)))  float  v8f;

#define WMMA_BF16(A, B, C) \
    __builtin_amdgcn_wmma_f32_16x16x32_bf16(false, (A), false, (B), (short)0, (C), false, false)

// Problem dims
#define BB 64
#define SS 512
#define II 256
#define HH 512
#define G3 1536      // 3*H
#define OO2 512      // 2*O

// ---------------------------------------------------------------------------
// Fragment loaders (wave32), contiguous K order used identically for A and B.
// A (16x32, MxK, 16-bit): lane L holds row M = L&15; lanes 0-15 carry
//   K = 0..15, lanes 16-31 carry K = 16..31.
// B (32x16, KxN, 16-bit): lane L holds col N = L&15; lanes 0-15 carry
//   K = 0..15, lanes 16-31 carry K = 16..31 (contiguous in K for a [N][K]
//   row-major weight matrix, i.e. exactly the PyTorch weight layout).
// ---------------------------------------------------------------------------
__device__ inline v16bf load_a_frag(const bf16_t* __restrict__ base, int ld,
                                    int row0, int k0, int lane) {
    const bf16_t* p = base + (size_t)(row0 + (lane & 15)) * ld + k0 + ((lane >> 4) << 4);
    v8bf lo = *(const v8bf*)(p);
    v8bf hi = *(const v8bf*)(p + 8);
    return __builtin_shufflevector(lo, hi, 0, 1, 2, 3, 4, 5, 6, 7,
                                           8, 9, 10, 11, 12, 13, 14, 15);
}

__device__ inline v16bf load_b_frag(const bf16_t* __restrict__ w, int ld,
                                    int n0, int k0, int lane) {
    const bf16_t* p = w + (size_t)(n0 + (lane & 15)) * ld + k0 + ((lane >> 4) << 4);
    return *(const v16bf*)(p);
}

// B fragment out of an LDS-staged [rows][512] slab (rows = g*16 + rr).
__device__ inline v16bf lds_b_frag(const bf16_t* slab, int g, int k0, int lane) {
    return *(const v16bf*)(slab + (g * 16 + (lane & 15)) * HH + k0 + ((lane >> 4) << 4));
}

// ---------------------------------------------------------------------------
// Setup kernels: zero state / downcast weights / transpose+downcast input.
// ---------------------------------------------------------------------------
__global__ void k_init(bf16_t* __restrict__ h0, float* __restrict__ hf,
                       unsigned* __restrict__ bar) {
    int i = blockIdx.x * blockDim.x + threadIdx.x;
    if (i == 0) *bar = 0u;
    if (i < BB * HH) { h0[i] = (bf16_t)0.0f; hf[i] = 0.0f; }
}

__global__ void k_convert(const float* __restrict__ in, bf16_t* __restrict__ out, int n) {
    int i = blockIdx.x * blockDim.x + threadIdx.x;
    if (i < n) out[i] = (bf16_t)in[i];
}

// input_var [B,S,I] f32  ->  xt [S*B, I] bf16 with row m = s*B + b
__global__ void k_xt(const float* __restrict__ x, bf16_t* __restrict__ xt) {
    int i = blockIdx.x * blockDim.x + threadIdx.x;   // over B*S*I = 8388608
    if (i < BB * SS * II) {
        int ii = i & (II - 1);
        int s  = (i >> 8) & (SS - 1);
        int b  = i >> 17;
        xt[(((size_t)s * BB + b) << 8) + ii] = (bf16_t)x[i];
    }
}

// ---------------------------------------------------------------------------
// gi = xt @ w_ih^T + b_ih   : M=32768, N=1536, K=256.  One 16x64 tile/wave,
// software-pipelined one k-step ahead.
// ---------------------------------------------------------------------------
__global__ __launch_bounds__(256) void k_gemm_gi(const bf16_t* __restrict__ xt,
                                                 const bf16_t* __restrict__ wih,
                                                 const float* __restrict__ bih,
                                                 float* __restrict__ gi) {
    int lane = threadIdx.x & 31;
    int gw   = (blockIdx.x * blockDim.x + threadIdx.x) >> 5;   // [0, 49152)
    int mt = gw & 2047;          // 2048 M-tiles
    int ng = gw >> 11;           // 24 N-groups of 64
    int m0 = mt << 4, n0 = ng << 6;

    v8f acc[4] = {};
    v16bf a_cur = load_a_frag(xt, II, m0, 0, lane);
    v16bf b_cur[4];
#pragma unroll
    for (int t = 0; t < 4; ++t) b_cur[t] = load_b_frag(wih, II, n0 + t * 16, 0, lane);

#pragma unroll
    for (int k0 = 32; k0 < II; k0 += 32) {
        v16bf a_nxt = load_a_frag(xt, II, m0, k0, lane);
        v16bf b_nxt[4];
#pragma unroll
        for (int t = 0; t < 4; ++t) b_nxt[t] = load_b_frag(wih, II, n0 + t * 16, k0, lane);
#pragma unroll
        for (int t = 0; t < 4; ++t) acc[t] = WMMA_BF16(a_cur, b_cur[t], acc[t]);
        a_cur = a_nxt;
#pragma unroll
        for (int t = 0; t < 4; ++t) b_cur[t] = b_nxt[t];
    }
#pragma unroll
    for (int t = 0; t < 4; ++t) acc[t] = WMMA_BF16(a_cur, b_cur[t], acc[t]);

    int mbase = m0 + ((lane >> 4) << 3);
#pragma unroll
    for (int t = 0; t < 4; ++t) {
        int n = n0 + t * 16 + (lane & 15);
        float bias = bih[n];
#pragma unroll
        for (int r = 0; r < 8; ++r)
            gi[(size_t)(mbase + r) * G3 + n] = acc[t][r] + bias;
    }
}

// ---------------------------------------------------------------------------
// Persistent GRU recurrence. 32 blocks x 4 waves = 128 waves = (4 b-tiles) x
// (32 j-tiles); block b owns j-tile b, its 4 waves cover the 4 batch tiles.
// The block's loop-invariant w_hh slice (3 gates x 16 rows x K=512, 48 KB)
// is DMA'd into LDS ONCE via async-to-LDS, then every step reads B from LDS
// (ds_load) while only the step-varying h_{s-1} comes from global memory
// (prefetched one k-step ahead). Gate operands (gi, h_prev) are fetched
// before the GEMM chain so their latency hides under the 48 WMMAs.
// One agent-scope atomic grid barrier per step orders h_s across blocks.
// ---------------------------------------------------------------------------
__global__ __launch_bounds__(128) void k_gru(const bf16_t* __restrict__ whh,
                                             const float*  __restrict__ bhh,
                                             const float*  __restrict__ gi,
                                             const bf16_t* __restrict__ h0,
                                             bf16_t* __restrict__ hs,
                                             float*  __restrict__ hf,
                                             unsigned* __restrict__ bar) {
    __shared__ bf16_t wlds[3 * 16 * HH];          // 48 KB: [g*16+rr][k]

    int lane = threadIdx.x & 31;
    int w    = threadIdx.x >> 5;                  // [0,4): batch tile
    int b0 = w << 4;
    int j0 = blockIdx.x << 4;                     // j-tile = blockIdx.x
    int jcol  = j0 + (lane & 15);
    int mbase = b0 + ((lane >> 4) << 3);
    float bh_r = bhh[jcol];
    float bh_z = bhh[HH + jcol];
    float bh_n = bhh[2 * HH + jcol];
    unsigned nblk = gridDim.x;

    // ---- one-time async DMA stage of w_hh slice into LDS -------------------
    // 48 rows x 1024 B = 3072 x 16 B chunks, 24 per thread.
    for (int c = threadIdx.x; c < 3072; c += 128) {
        int row  = c >> 6;                        // [0,48) = g*16 + rr
        int koff = (c & 63) << 3;                 // element offset, 8 per 16B
        int g  = row >> 4;
        int rr = row & 15;
        const bf16_t* gsrc = whh + (size_t)(g * HH + j0 + rr) * HH + koff;
        unsigned ldst = (unsigned)(size_t)(const void*)&wlds[row * HH + koff];
        asm volatile("global_load_async_to_lds_b128 %0, %1, off"
                     :: "v"(ldst), "v"(gsrc) : "memory");
    }
    asm volatile("s_wait_asynccnt 0x0" ::: "memory");
    __syncthreads();

    for (int s = 0; s < SS; ++s) {
        const bf16_t* hprev = (s == 0) ? h0 : hs + (size_t)(s - 1) * BB * HH;
        const float*  gis   = gi + (size_t)s * BB * G3;
        bf16_t*       hrow  = hs + (size_t)s * BB * HH;

        // Prefetch gate operands: latency overlaps the whole WMMA chain.
        float gir[8], giz[8], gin[8], hp[8];
#pragma unroll
        for (int r = 0; r < 8; ++r) {
            int b = mbase + r;
            gir[r] = gis[(size_t)b * G3 + jcol];
            giz[r] = gis[(size_t)b * G3 + HH + jcol];
            gin[r] = gis[(size_t)b * G3 + 2 * HH + jcol];
            hp[r]  = hf[b * HH + jcol];
        }

        v8f ar = {}, az = {}, an = {};
        v16bf a_cur = load_a_frag(hprev, HH, b0, 0, lane);
#pragma unroll
        for (int k0 = 0; k0 < HH - 32; k0 += 32) {
            v16bf a_nxt = load_a_frag(hprev, HH, b0, k0 + 32, lane);
            v16bf br = lds_b_frag(wlds, 0, k0, lane);
            v16bf bz = lds_b_frag(wlds, 1, k0, lane);
            v16bf bn = lds_b_frag(wlds, 2, k0, lane);
            ar = WMMA_BF16(a_cur, br, ar);
            az = WMMA_BF16(a_cur, bz, az);
            an = WMMA_BF16(a_cur, bn, an);
            a_cur = a_nxt;
        }
        {   // peeled final k-step
            v16bf br = lds_b_frag(wlds, 0, HH - 32, lane);
            v16bf bz = lds_b_frag(wlds, 1, HH - 32, lane);
            v16bf bn = lds_b_frag(wlds, 2, HH - 32, lane);
            ar = WMMA_BF16(a_cur, br, ar);
            az = WMMA_BF16(a_cur, bz, az);
            an = WMMA_BF16(a_cur, bn, an);
        }

#pragma unroll
        for (int r = 0; r < 8; ++r) {
            int b = mbase + r;
            float rg  = 1.0f / (1.0f + __expf(-(gir[r] + ar[r] + bh_r)));
            float zg  = 1.0f / (1.0f + __expf(-(giz[r] + az[r] + bh_z)));
            float ngt = tanhf(gin[r] + rg * (an[r] + bh_n));
            float hn  = (1.0f - zg) * ngt + zg * hp[r];
            hf[b * HH + jcol] = hn;
            hrow[(size_t)b * HH + jcol] = (bf16_t)hn;
        }

        // ---- device-wide barrier: release, arrive, spin, acquire ----
        __builtin_amdgcn_fence(__ATOMIC_RELEASE, "agent");
        __syncthreads();
        if (threadIdx.x == 0) {
            __hip_atomic_fetch_add(bar, 1u, __ATOMIC_ACQ_REL, __HIP_MEMORY_SCOPE_AGENT);
            unsigned target = nblk * (unsigned)(s + 1);
            while (__hip_atomic_load(bar, __ATOMIC_ACQUIRE, __HIP_MEMORY_SCOPE_AGENT) < target)
                __builtin_amdgcn_s_sleep(8);
        }
        __syncthreads();
        __builtin_amdgcn_fence(__ATOMIC_ACQUIRE, "agent");
    }
}

// ---------------------------------------------------------------------------
// out = hs @ w_out^T + b_out : M=32768 (m = s*B+b), N=512, K=512.
// d_out layout [B,S,O,2]: element (m,n) -> d_out[(b*S + s)*512 + n].
// Software-pipelined one k-step ahead.
// ---------------------------------------------------------------------------
__global__ __launch_bounds__(256) void k_gemm_out(const bf16_t* __restrict__ hs,
                                                  const bf16_t* __restrict__ wout,
                                                  const float* __restrict__ bout,
                                                  float* __restrict__ out) {
    int lane = threadIdx.x & 31;
    int gw   = (blockIdx.x * blockDim.x + threadIdx.x) >> 5;   // [0, 16384)
    int mt = gw & 2047;
    int ng = gw >> 11;            // [0, 8)
    int m0 = mt << 4, n0 = ng << 6;

    v8f acc[4] = {};
    v16bf a_cur = load_a_frag(hs, HH, m0, 0, lane);
    v16bf b_cur[4];
#pragma unroll
    for (int t = 0; t < 4; ++t) b_cur[t] = load_b_frag(wout, HH, n0 + t * 16, 0, lane);

#pragma unroll
    for (int k0 = 32; k0 < HH; k0 += 32) {
        v16bf a_nxt = load_a_frag(hs, HH, m0, k0, lane);
        v16bf b_nxt[4];
#pragma unroll
        for (int t = 0; t < 4; ++t) b_nxt[t] = load_b_frag(wout, HH, n0 + t * 16, k0, lane);
#pragma unroll
        for (int t = 0; t < 4; ++t) acc[t] = WMMA_BF16(a_cur, b_cur[t], acc[t]);
        a_cur = a_nxt;
#pragma unroll
        for (int t = 0; t < 4; ++t) b_cur[t] = b_nxt[t];
    }
#pragma unroll
    for (int t = 0; t < 4; ++t) acc[t] = WMMA_BF16(a_cur, b_cur[t], acc[t]);

    int mbase = m0 + ((lane >> 4) << 3);
#pragma unroll
    for (int t = 0; t < 4; ++t) {
        int n = n0 + t * 16 + (lane & 15);
        float bias = bout[n];
#pragma unroll
        for (int r = 0; r < 8; ++r) {
            int m = mbase + r;
            int b = m & (BB - 1);
            int s = m >> 6;
            out[((size_t)b * SS + s) * OO2 + n] = acc[t][r] + bias;
        }
    }
}

// ---------------------------------------------------------------------------
// Host launcher
// ---------------------------------------------------------------------------
extern "C" void kernel_launch(void* const* d_in, const int* in_sizes, int n_in,
                              void* d_out, int out_size, void* d_ws, size_t ws_size,
                              hipStream_t stream) {
    (void)in_sizes; (void)n_in; (void)out_size; (void)ws_size;
    const float* x    = (const float*)d_in[0];   // [B,S,I]
    const float* wih  = (const float*)d_in[1];   // [3H,I]
    const float* whh  = (const float*)d_in[2];   // [3H,H]
    const float* bih  = (const float*)d_in[3];   // [3H]
    const float* bhh  = (const float*)d_in[4];   // [3H]
    const float* wout = (const float*)d_in[5];   // [2O,H]
    const float* bout = (const float*)d_in[6];   // [2O]
    float* out = (float*)d_out;                  // [B,S,O,2]

    char* ws = (char*)d_ws;
    size_t off = 0;
    auto alloc = [&](size_t bytes) -> void* {
        void* p = ws + off;
        off += (bytes + 255) & ~(size_t)255;
        return p;
    };
    bf16_t*   xt    = (bf16_t*)alloc((size_t)SS * BB * II * 2);   //  16.8 MB
    bf16_t*   wihb  = (bf16_t*)alloc((size_t)G3 * II * 2);        //   0.8 MB
    bf16_t*   whhb  = (bf16_t*)alloc((size_t)G3 * HH * 2);        //   1.6 MB
    bf16_t*   woutb = (bf16_t*)alloc((size_t)OO2 * HH * 2);       //   0.5 MB
    float*    gi    = (float*) alloc((size_t)SS * BB * G3 * 4);   // 201.3 MB
    bf16_t*   hs    = (bf16_t*)alloc((size_t)SS * BB * HH * 2);   //  33.6 MB
    bf16_t*   h0b   = (bf16_t*)alloc((size_t)BB * HH * 2);
    float*    hf    = (float*) alloc((size_t)BB * HH * 4);
    unsigned* bar   = (unsigned*)alloc(256);

    // 1) init state + downcast weights + transpose/downcast input
    k_init<<<(BB * HH + 255) / 256, 256, 0, stream>>>(h0b, hf, bar);
    k_xt<<<(BB * SS * II + 255) / 256, 256, 0, stream>>>(x, xt);
    k_convert<<<(G3 * II + 255) / 256, 256, 0, stream>>>(wih, wihb, G3 * II);
    k_convert<<<(G3 * HH + 255) / 256, 256, 0, stream>>>(whh, whhb, G3 * HH);
    k_convert<<<(OO2 * HH + 255) / 256, 256, 0, stream>>>(wout, woutb, OO2 * HH);

    // 2) bulk input-side gate projections (WMMA, double-buffered)
    k_gemm_gi<<<6144, 256, 0, stream>>>(xt, wihb, bih, gi);

    // 3) persistent sequential recurrence (WMMA, LDS-staged weights,
    //    async-to-LDS DMA, prefetch pipelining, grid barrier per step)
    k_gru<<<32, 128, 0, stream>>>(whhb, bhh, gi, h0b, hs, hf, bar);

    // 4) bulk output projection (WMMA, double-buffered)
    k_gemm_out<<<2048, 256, 0, stream>>>(hs, woutb, bout, out);
}
